// SelfAttention_80444737454804
// MI455X (gfx1250) — matrix operand order repaired and checked by reading, not compiled, not run on hardware
//
#include <hip/hip_runtime.h>

typedef __bf16 bf16;
typedef __attribute__((ext_vector_type(16))) __bf16 v16bf;
typedef __attribute__((ext_vector_type(8)))  float  v8f;

#define SCALE 0.03125f            // 1024^-0.5 (reference scales by full dim)
#define NEG_INF (-__builtin_huge_valf())

__device__ __forceinline__ v8f wmma_bf16(v16bf a, v16bf b, v8f c) {
  // (neg_a, A, neg_b, B, c_mod, C, reuse_a, reuse_b)
  return __builtin_amdgcn_wmma_f32_16x16x32_bf16(false, a, false, b, (short)0, c, false, false);
}

// A-matrix fragment (16x32 bf16). lane=(half,r): row M=r.
// elements 0-7  <-> K = half*8 + 0..7 ; elements 8-15 <-> K = 16 + half*8 + 0..7
__device__ __forceinline__ v16bf fragA(const bf16* rowp, int half) {
  v16bf f;
  ((uint4*)&f)[0] = *(const uint4*)(rowp + half * 8);
  ((uint4*)&f)[1] = *(const uint4*)(rowp + 16 + half * 8);
  return f;
}

// B-matrix fragment (32x16 bf16) from Bt storage (row n = B[:,n]).
// lane=(half,r): col N=r; elements 0-7 <-> K = half*8 + 0..7 ; elements 8-15 <-> K = 16 + half*8 + 0..7
__device__ __forceinline__ v16bf fragB(const bf16* colp, int half) {
  v16bf f;
  ((uint4*)&f)[0] = *(const uint4*)(colp + half * 8);
  ((uint4*)&f)[1] = *(const uint4*)(colp + 16 + half * 8);
  return f;
}

// ---------------- Kernel 1: QKV projection  X[4096,1024] @ W[1024,3072] ----------------
__global__ __launch_bounds__(256) void qkv_gemm_kernel(
    const float* __restrict__ X, const float* __restrict__ W,
    bf16* __restrict__ Qw, bf16* __restrict__ Kw, bf16* __restrict__ Vtw) {
  __shared__ bf16 As[128][40];
  __shared__ bf16 Bs[128][40];
  const int tid = threadIdx.x;
  const int lane = tid & 31, wid = tid >> 5;
  const int half = lane >> 4, r = lane & 15;
  const int wm = wid >> 2, wn = wid & 3;              // 2x4 wave grid
  const int m0 = blockIdx.y * 128, n0 = blockIdx.x * 128;
  const int K = 1024, N = 3072;

  v8f z = {0.f, 0.f, 0.f, 0.f, 0.f, 0.f, 0.f, 0.f};
  v8f acc[4][2];
  for (int mf = 0; mf < 4; ++mf)
    for (int nf = 0; nf < 2; ++nf) acc[mf][nf] = z;

  const int arow = tid >> 3, acol = (tid & 7) << 2;
  const int brow = tid >> 5, bcol = (tid & 31) << 2;

  for (int k0 = 0; k0 < K; k0 += 32) {
#pragma unroll
    for (int p = 0; p < 4; ++p) {
      float4 v = *(const float4*)(X + (size_t)(m0 + arow + p * 32) * K + (k0 + acol));
      bf16* d = &As[arow + p * 32][acol];
      d[0] = (bf16)v.x; d[1] = (bf16)v.y; d[2] = (bf16)v.z; d[3] = (bf16)v.w;
    }
#pragma unroll
    for (int p = 0; p < 4; ++p) {
      float4 v = *(const float4*)(W + (size_t)(k0 + brow + p * 8) * N + (n0 + bcol));
      int kk = brow + p * 8;
      Bs[bcol + 0][kk] = (bf16)v.x; Bs[bcol + 1][kk] = (bf16)v.y;
      Bs[bcol + 2][kk] = (bf16)v.z; Bs[bcol + 3][kk] = (bf16)v.w;
    }
    __syncthreads();
    v16bf af[4], bfr[2];
#pragma unroll
    for (int mf = 0; mf < 4; ++mf) af[mf] = fragA(&As[wm * 64 + mf * 16 + r][0], half);
#pragma unroll
    for (int nf = 0; nf < 2; ++nf) bfr[nf] = fragB(&Bs[wn * 32 + nf * 16 + r][0], half);
#pragma unroll
    for (int mf = 0; mf < 4; ++mf)
#pragma unroll
      for (int nf = 0; nf < 2; ++nf)
        acc[mf][nf] = wmma_bf16(af[mf], bfr[nf], acc[mf][nf]);
    __syncthreads();
  }

  // scatter into Q (pre-scaled), K, V^T  (per b,h layouts)
#pragma unroll
  for (int mf = 0; mf < 4; ++mf)
#pragma unroll
    for (int nf = 0; nf < 2; ++nf)
#pragma unroll
      for (int i = 0; i < 8; ++i) {
        int gm = m0 + wm * 64 + mf * 16 + half * 8 + i;
        int gc = n0 + wn * 32 + nf * 16 + r;
        float val = acc[mf][nf][i];
        int b = gm >> 11, nrow = gm & 2047;
        int part = gc >> 10, rest = gc & 1023;
        int head = rest >> 6, d = rest & 63;
        size_t base = (size_t)(b * 16 + head) * (2048 * 64);
        if (part == 0)      Qw[base + (size_t)nrow * 64 + d] = (bf16)(val * SCALE);
        else if (part == 1) Kw[base + (size_t)nrow * 64 + d] = (bf16)val;
        else                Vtw[base + (size_t)d * 2048 + nrow] = (bf16)val;
      }
}

// ---------------- Kernel 2: causal flash attention per (b,h) ----------------
__global__ __launch_bounds__(256) void attn_kernel(
    const bf16* __restrict__ Qw, const bf16* __restrict__ Kw,
    const bf16* __restrict__ Vtw, bf16* __restrict__ Ow) {
  __shared__ bf16 Pl[8][16][40];                       // wave-private P tiles
  const int tid = threadIdx.x, lane = tid & 31, wid = tid >> 5;
  const int half = lane >> 4, r = lane & 15;
  const int bh = blockIdx.y;
  const int m_base = blockIdx.x * 128 + wid * 16;      // 16 query rows per wave
  const bf16* Qp = Qw + (size_t)bh * 2048 * 64;
  const bf16* Kp = Kw + (size_t)bh * 2048 * 64;
  const bf16* Vp = Vtw + (size_t)bh * 64 * 2048;

  v16bf qf0 = fragA(Qp + (m_base + r) * 64, half);       // d = 0..31
  v16bf qf1 = fragA(Qp + (m_base + r) * 64 + 32, half);  // d = 32..63

  v8f z = {0.f, 0.f, 0.f, 0.f, 0.f, 0.f, 0.f, 0.f};
  v8f o[4] = {z, z, z, z};
  float mrow[8], lrow[8];
#pragma unroll
  for (int i = 0; i < 8; ++i) { mrow[i] = NEG_INF; lrow[i] = 0.f; }

  for (int jb = 0; jb <= m_base + 15; jb += 32) {       // causal block range
    v8f s0 = z, s1 = z;
    {
      const bf16* k0p = Kp + (jb + r) * 64;
      s0 = wmma_bf16(qf0, fragB(k0p, half), s0);
      s0 = wmma_bf16(qf1, fragB(k0p + 32, half), s0);
      const bf16* k1p = Kp + (jb + 16 + r) * 64;
      s1 = wmma_bf16(qf0, fragB(k1p, half), s1);
      s1 = wmma_bf16(qf1, fragB(k1p + 32, half), s1);
    }
    // causal mask + online softmax (rows i live in lanes of this half)
#pragma unroll
    for (int i = 0; i < 8; ++i) {
      int row = m_base + half * 8 + i;
      float a0 = s0[i], a1 = s1[i];
      if (jb + r > row)      a0 = NEG_INF;
      if (jb + 16 + r > row) a1 = NEG_INF;
      float mx = fmaxf(a0, a1);
      mx = fmaxf(mx, __shfl_xor(mx, 1, 32));
      mx = fmaxf(mx, __shfl_xor(mx, 2, 32));
      mx = fmaxf(mx, __shfl_xor(mx, 4, 32));
      mx = fmaxf(mx, __shfl_xor(mx, 8, 32));
      float nm = fmaxf(mrow[i], mx);
      float p0 = __expf(a0 - nm), p1 = __expf(a1 - nm);
      float sum = p0 + p1;
      sum += __shfl_xor(sum, 1, 32);
      sum += __shfl_xor(sum, 2, 32);
      sum += __shfl_xor(sum, 4, 32);
      sum += __shfl_xor(sum, 8, 32);
      float corr = __expf(mrow[i] - nm);
      lrow[i] = lrow[i] * corr + sum;
      mrow[i] = nm;
#pragma unroll
      for (int t = 0; t < 4; ++t) o[t][i] = o[t][i] * corr;
      Pl[wid][half * 8 + i][r] = (bf16)p0;             // C-layout -> LDS
      Pl[wid][half * 8 + i][16 + r] = (bf16)p1;
    }
    // wave-private LDS round trip converts C layout -> A fragment (same-wave DS ops are in order)
    v16bf pf = fragA(&Pl[wid][r][0], half);
#pragma unroll
    for (int t = 0; t < 4; ++t) {
      v16bf vf = fragB(Vp + (size_t)(t * 16 + r) * 2048 + jb, half);
      o[t] = wmma_bf16(pf, vf, o[t]);
    }
  }

  const int h = bh & 15, bb = bh >> 4;
#pragma unroll
  for (int i = 0; i < 8; ++i) {
    float inv = 1.0f / lrow[i];
    size_t rowoff = (size_t)(bb * 2048 + m_base + half * 8 + i) * 1024 + h * 64;
#pragma unroll
    for (int t = 0; t < 4; ++t)
      Ow[rowoff + t * 16 + r] = (bf16)(o[t][i] * inv);
  }
}

// ---------------- Kernel 3: output projection O[4096,1024] @ Wout[1024,1024] + b ----------------
__global__ __launch_bounds__(256) void out_gemm_kernel(
    const bf16* __restrict__ A, const float* __restrict__ W,
    const float* __restrict__ bias, float* __restrict__ out) {
  __shared__ bf16 As[128][40];
  __shared__ bf16 Bs[128][40];
  const int tid = threadIdx.x;
  const int lane = tid & 31, wid = tid >> 5;
  const int half = lane >> 4, r = lane & 15;
  const int wm = wid >> 2, wn = wid & 3;
  const int m0 = blockIdx.y * 128, n0 = blockIdx.x * 128;
  const int K = 1024, N = 1024;

  v8f z = {0.f, 0.f, 0.f, 0.f, 0.f, 0.f, 0.f, 0.f};
  v8f acc[4][2];
  for (int mf = 0; mf < 4; ++mf)
    for (int nf = 0; nf < 2; ++nf) acc[mf][nf] = z;

  const int arow = tid >> 1, aseg = (tid & 1) * 16;
  const int brow = tid >> 5, bcol = (tid & 31) << 2;

  for (int k0 = 0; k0 < K; k0 += 32) {
    {  // A already bf16: straight 32B copy per thread
      const bf16* src = A + (size_t)(m0 + arow) * K + k0 + aseg;
      *(uint4*)&As[arow][aseg] = ((const uint4*)src)[0];
      *(uint4*)&As[arow][aseg + 8] = ((const uint4*)src)[1];
    }
#pragma unroll
    for (int p = 0; p < 4; ++p) {
      float4 v = *(const float4*)(W + (size_t)(k0 + brow + p * 8) * N + (n0 + bcol));
      int kk = brow + p * 8;
      Bs[bcol + 0][kk] = (bf16)v.x; Bs[bcol + 1][kk] = (bf16)v.y;
      Bs[bcol + 2][kk] = (bf16)v.z; Bs[bcol + 3][kk] = (bf16)v.w;
    }
    __syncthreads();
    v16bf af[4], bfr[2];
#pragma unroll
    for (int mf = 0; mf < 4; ++mf) af[mf] = fragA(&As[wm * 64 + mf * 16 + r][0], half);
#pragma unroll
    for (int nf = 0; nf < 2; ++nf) bfr[nf] = fragB(&Bs[wn * 32 + nf * 16 + r][0], half);
#pragma unroll
    for (int mf = 0; mf < 4; ++mf)
#pragma unroll
      for (int nf = 0; nf < 2; ++nf)
        acc[mf][nf] = wmma_bf16(af[mf], bfr[nf], acc[mf][nf]);
    __syncthreads();
  }

#pragma unroll
  for (int mf = 0; mf < 4; ++mf)
#pragma unroll
    for (int nf = 0; nf < 2; ++nf)
#pragma unroll
      for (int i = 0; i < 8; ++i) {
        int gm = m0 + wm * 64 + mf * 16 + half * 8 + i;
        int gc = n0 + wn * 32 + nf * 16 + r;
        out[(size_t)gm * N + gc] = acc[mf][nf][i] + bias[gc];
      }
}

extern "C" void kernel_launch(void* const* d_in, const int* in_sizes, int n_in,
                              void* d_out, int out_size, void* d_ws, size_t ws_size,
                              hipStream_t stream) {
  const float* x     = (const float*)d_in[0];
  const float* w_qkv = (const float*)d_in[1];
  const float* w_out = (const float*)d_in[2];
  const float* b_out = (const float*)d_in[3];
  float* out = (float*)d_out;

  // workspace layout (bf16): Q | K | V^T | attn-out, each 2*16*2048*64 = 4M elems (8MB)
  bf16* Qw  = (bf16*)d_ws;
  bf16* Kw  = Qw  + (size_t)4194304;
  bf16* Vtw = Kw  + (size_t)4194304;
  bf16* Ow  = Vtw + (size_t)4194304;

  dim3 blk(256);
  qkv_gemm_kernel<<<dim3(24, 32), blk, 0, stream>>>(x, w_qkv, Qw, Kw, Vtw);
  attn_kernel<<<dim3(16, 32), blk, 0, stream>>>(Qw, Kw, Vtw, Ow);
  out_gemm_kernel<<<dim3(8, 32), blk, 0, stream>>>(Ow, w_out, b_out, out);
}
